// Dynamics_49143015801545
// MI455X (gfx1250) — matrix operand order repaired and checked by reading, not compiled, not run on hardware
//
#include <hip/hip_runtime.h>
#include <hip/hip_bf16.h>
#include <math.h>

// ---------------------------------------------------------------------------
// Problem constants
// ---------------------------------------------------------------------------
#define NB  2048   // batch
#define OO  16     // objects
#define CLC 64     // channels

typedef __attribute__((ext_vector_type(16))) _Float16 v16h;
typedef __attribute__((ext_vector_type(8)))  float    v8f;

// ---------------------------------------------------------------------------
// f16 weight arena: per-dynamics offsets (in halves), row-major [out][in]
// ---------------------------------------------------------------------------
#define OFF_RELW1  0        // 128x128 (col 128 of the 129 handled as rank-1 f32)
#define OFF_ATTW1  16384    // 128x128
#define OFF_RELW2  32768    // 64x128
#define OFF_ATTW2  40960    // 64x128
#define OFF_RELW3  49152    // 64x64
#define OFF_SELFW1 53248    // 64x64
#define OFF_SELFW2 57344    // 64x64
#define OFF_AFFW1  61440    // 64x64
#define OFF_AFFW2  65536    // 64x64
#define OFF_AFFW3  69632    // 64x64
#define OFF_OUTW1  73728    // 128x128
#define OFF_OUTW2  90112    // 128x128
#define PERD       106496   // halves per dynamics

// ---------------------------------------------------------------------------
// workspace byte offsets
// ---------------------------------------------------------------------------
#define WS_S2   0ull               // N*O*64 f32   = 8388608 B
#define WS_DYNT 8388608ull         // N*O*3  f32   =  393216 B
#define WS_DPL  8781824ull         // N*O*64 f32   = 8388608 B
#define WS_WH   17170432ull        // 3*PERD halves =  638976 B

// ---------------------------------------------------------------------------
// K2 dynamic-LDS layout (byte offsets)
// ---------------------------------------------------------------------------
#define L_COMB 0         // 256x128 f16 = 65536
#define L_BUF1 65536     // 256x128 f16 = 65536 (r1 / t1)
#define L_BUF2 131072    // 256x64  f16 = 32768 (r2 / t2)
#define L_BUF3 163840    // 256x64  f16 = 32768 (rel3)
#define L_WST  196608    // staged weights, up to 128x128 f16 = 32768
#define L_S2H  229376    // 16x64 f16 = 2048
#define L_DIST 231424    // 256 f32   = 1024
#define L_ATT  232448    // 256 f32   = 1024
#define L_RELD 233472    // 16x64 f32 = 4096
#define L_H1H  237568    // 16x64 f16 = 2048
#define L_HF   239616    // 16x64 f32 = 4096
#define L_DPH  243712    // 16x64 f16 = 2048
#define L_DPF  245760    // 16x64 f32 = 4096
#define L_F1H  249856    // 16x64 f16 = 2048
#define L_F1F  251904    // 16x64 f32 = 4096
#define L_F2H  256000    // 16x64 f16 = 2048
#define L_OUTA 258048    // 16x128 f16 = 4096
#define L_O1H  262144    // 16x128 f16 = 4096
#define L_O1F  266240    // 16x128 f32 = 8192
#define L_RES  274432    // 16x128 f32 = 8192
#define L_DYNT 282624    // 48 f32 (pad 256)
#define L_PX   282880    // 16 f32
#define L_PY   282944    // 16 f32
#define SMEM_TOTAL 283008

// ---------------------------------------------------------------------------
// WMMA fragment helpers (CDNA5 16x16x32 f16, wave32 — ISA 7.12.2 layouts)
//   A (16x32, M x K): lane<16 holds K {0..7,16..23}, lane>=16 holds {8..15,24..31}
//   B (32x16, K x N): lane holds column n=lane&15; same K order as A:
//                     lane<16 K {0..7,16..23}, lane>=16 {8..15,24..31}  (row-major [n][k] weight => B)
//   D (16x16 f32):    lane col n=lane&15, row m = r + 8*(lane>=16)
// ---------------------------------------------------------------------------
__device__ __forceinline__ v16h ldfragA(const _Float16* p, int lh) {
  p += lh * 8;
  v16h r;
#pragma unroll
  for (int i = 0; i < 8; ++i) r[i] = p[i];
#pragma unroll
  for (int i = 0; i < 8; ++i) r[8 + i] = p[16 + i];
  return r;
}

__device__ __forceinline__ v16h ldfragB(const _Float16* p, int lh) {
  p += lh * 8;
  v16h r;
#pragma unroll
  for (int i = 0; i < 8; ++i) r[i] = p[i];
#pragma unroll
  for (int i = 0; i < 8; ++i) r[8 + i] = p[16 + i];
  return r;
}

__device__ __forceinline__ v8f wmma_f16(v16h a, v16h b, v8f c) {
  return __builtin_amdgcn_wmma_f32_16x16x32_f16(false, a, false, b, (short)0, c,
                                                false, false);
}

__device__ __forceinline__ void stageW(_Float16* dst, const _Float16* __restrict__ src,
                                       int nHalves, int t) {
  uint32_t* d = (uint32_t*)dst;
  const uint32_t* s = (const uint32_t*)src;
  int n32 = nHalves >> 1;
  for (int p = t; p < n32; p += 512) d[p] = s[p];
}

// edge GEMM: 256xK in LDS @ W(NxK in LDS).T ; one M-tile per wave (16 waves)
__device__ __forceinline__ void edge_gemm1(  // K=128(+rank1 dist), N=128, relu
    const _Float16* inb, const _Float16* wst, const float* __restrict__ bias,
    const float* __restrict__ wfull129, const float* distA, _Float16* outb,
    int wv, int lr, int lh, int mh) {
  const _Float16* arow = inb + (wv * 16 + lr) * 128;
  v16h a0 = ldfragA(arow, lh), a1 = ldfragA(arow + 32, lh);
  v16h a2 = ldfragA(arow + 64, lh), a3 = ldfragA(arow + 96, lh);
#pragma unroll
  for (int nt = 0; nt < 8; ++nt) {
    int col = nt * 16 + lr;
    float bv = bias[col];
    v8f c;
#pragma unroll
    for (int r = 0; r < 8; ++r) c[r] = bv;
    const _Float16* wrow = wst + col * 128;
    c = wmma_f16(a0, ldfragB(wrow, lh), c);
    c = wmma_f16(a1, ldfragB(wrow + 32, lh), c);
    c = wmma_f16(a2, ldfragB(wrow + 64, lh), c);
    c = wmma_f16(a3, ldfragB(wrow + 96, lh), c);
    float wl = wfull129[col * 129 + 128];  // the "+dist" column, kept f32
#pragma unroll
    for (int r = 0; r < 8; ++r) {
      int row = wv * 16 + mh + r;
      float v = c[r] + distA[row] * wl;
      outb[row * 128 + col] = (_Float16)fmaxf(v, 0.f);
    }
  }
}

__device__ __forceinline__ void edge_gemm2(  // K=128, N=64, relu
    const _Float16* inb, const _Float16* wst, const float* __restrict__ bias,
    _Float16* outb, int wv, int lr, int lh, int mh) {
  const _Float16* arow = inb + (wv * 16 + lr) * 128;
  v16h a0 = ldfragA(arow, lh), a1 = ldfragA(arow + 32, lh);
  v16h a2 = ldfragA(arow + 64, lh), a3 = ldfragA(arow + 96, lh);
#pragma unroll
  for (int nt = 0; nt < 4; ++nt) {
    int col = nt * 16 + lr;
    float bv = bias[col];
    v8f c;
#pragma unroll
    for (int r = 0; r < 8; ++r) c[r] = bv;
    const _Float16* wrow = wst + col * 128;
    c = wmma_f16(a0, ldfragB(wrow, lh), c);
    c = wmma_f16(a1, ldfragB(wrow + 32, lh), c);
    c = wmma_f16(a2, ldfragB(wrow + 64, lh), c);
    c = wmma_f16(a3, ldfragB(wrow + 96, lh), c);
#pragma unroll
    for (int r = 0; r < 8; ++r) {
      int row = wv * 16 + mh + r;
      outb[row * 64 + col] = (_Float16)fmaxf(c[r], 0.f);
    }
  }
}

__device__ __forceinline__ void edge_gemm3_res(  // K=64, N=64, + residual(in)
    const _Float16* inb, const _Float16* wst, const float* __restrict__ bias,
    _Float16* outb, int wv, int lr, int lh, int mh) {
  const _Float16* arow = inb + (wv * 16 + lr) * 64;
  v16h a0 = ldfragA(arow, lh), a1 = ldfragA(arow + 32, lh);
#pragma unroll
  for (int nt = 0; nt < 4; ++nt) {
    int col = nt * 16 + lr;
    float bv = bias[col];
    v8f c;
#pragma unroll
    for (int r = 0; r < 8; ++r) c[r] = bv;
    const _Float16* wrow = wst + col * 64;
    c = wmma_f16(a0, ldfragB(wrow, lh), c);
    c = wmma_f16(a1, ldfragB(wrow + 32, lh), c);
#pragma unroll
    for (int r = 0; r < 8; ++r) {
      int row = wv * 16 + mh + r;
      float v = c[r] + (float)inb[row * 64 + col];
      outb[row * 64 + col] = (_Float16)v;
    }
  }
}

// node GEMM: single 16-row M-tile; B-frags streamed from global f16 weights
__device__ __forceinline__ v8f node_gemm(const _Float16* A, int lda,
                                         const _Float16* __restrict__ Wg, int ldw,
                                         const float* __restrict__ bias, int nt,
                                         int ksteps, int lr, int lh) {
  int col = nt * 16 + lr;
  float bv = bias[col];
  v8f c;
#pragma unroll
  for (int r = 0; r < 8; ++r) c[r] = bv;
  const _Float16* arow = A + lr * lda;
  const _Float16* wrow = Wg + col * ldw;
  for (int ks = 0; ks < ksteps; ++ks)
    c = wmma_f16(ldfragA(arow + ks * 32, lh), ldfragB(wrow + ks * 32, lh), c);
  return c;
}

// ---------------------------------------------------------------------------
// K0: convert GEMM weights f32 -> f16 arena
// ---------------------------------------------------------------------------
__global__ void k0_convert(_Float16* __restrict__ wh,
    const float* __restrict__ relW1, const float* __restrict__ attW1,
    const float* __restrict__ relW2, const float* __restrict__ attW2,
    const float* __restrict__ relW3, const float* __restrict__ selfW1,
    const float* __restrict__ selfW2, const float* __restrict__ affW1,
    const float* __restrict__ affW2, const float* __restrict__ affW3,
    const float* __restrict__ outW1, const float* __restrict__ outW2) {
  int tdx = blockIdx.x * 256 + threadIdx.x;
  if (tdx >= 3 * PERD) return;
  int d = tdx / PERD, r = tdx % PERD;
  float v;
  if      (r < OFF_ATTW1)  { int q = r - OFF_RELW1;  v = relW1[(d*128 + (q>>7))*129 + (q&127)]; }
  else if (r < OFF_RELW2)  { int q = r - OFF_ATTW1;  v = attW1[(d*128 + (q>>7))*129 + (q&127)]; }
  else if (r < OFF_ATTW2)  { int q = r - OFF_RELW2;  v = relW2[(d*64  + (q>>7))*128 + (q&127)]; }
  else if (r < OFF_RELW3)  { int q = r - OFF_ATTW2;  v = attW2[(d*64  + (q>>7))*128 + (q&127)]; }
  else if (r < OFF_SELFW1) { int q = r - OFF_RELW3;  v = relW3[(d*64  + (q>>6))*64  + (q&63)];  }
  else if (r < OFF_SELFW2) { int q = r - OFF_SELFW1; v = selfW1[(d*64 + (q>>6))*64  + (q&63)];  }
  else if (r < OFF_AFFW1)  { int q = r - OFF_SELFW2; v = selfW2[(d*64 + (q>>6))*64  + (q&63)];  }
  else if (r < OFF_AFFW2)  { int q = r - OFF_AFFW1;  v = affW1[(d*64  + (q>>6))*64  + (q&63)];  }
  else if (r < OFF_AFFW3)  { int q = r - OFF_AFFW2;  v = affW2[(d*64  + (q>>6))*64  + (q&63)];  }
  else if (r < OFF_OUTW1)  { int q = r - OFF_AFFW3;  v = affW3[(d*64  + (q>>6))*64  + (q&63)];  }
  else if (r < OFF_OUTW2)  { int q = r - OFF_OUTW1;  v = outW1[(d*128 + (q>>7))*128 + (q&127)]; }
  else                     { int q = r - OFF_OUTW2;  v = outW2[(d*128 + (q>>7))*128 + (q&127)]; }
  wh[tdx] = (_Float16)v;
}

// ---------------------------------------------------------------------------
// K1: s2 + dyn_types (tiny, f32 VALU)
// ---------------------------------------------------------------------------
__global__ __launch_bounds__(256) void k1_node(
    const float* __restrict__ s, const float* __restrict__ actions,
    const float* __restrict__ actW, const float* __restrict__ actB,
    const float* __restrict__ seW, const float* __restrict__ seB,
    const float* __restrict__ dW1, const float* __restrict__ dB1,
    const float* __restrict__ dW2, const float* __restrict__ dB2,
    float* __restrict__ s2ws, float* __restrict__ dyntws) {
  __shared__ float sIn[1024], act[16], ae[64], s2l[1024], h1[1024], lg[48];
  int n = blockIdx.x, t = threadIdx.x;
  for (int p = t; p < 1024; p += 256) sIn[p] = s[(size_t)n * 1024 + p];
  if (t < 9) act[t] = actions[(size_t)n * 9 + t];
  __syncthreads();
  if (t < 64) {
    float a = actB[t];
    for (int k = 0; k < 9; ++k) a += act[k] * actW[t * 9 + k];
    ae[t] = a;
  }
  __syncthreads();
  for (int p = t; p < 1024; p += 256) {
    int o = p >> 6, c = p & 63;
    float v;
    if (c < 2) v = sIn[o * 64 + c];
    else {
      v = seB[c];
      for (int k = 0; k < 64; ++k) v += sIn[o * 64 + k] * seW[c * 68 + k];
      for (int k = 0; k < 4; ++k) v += ae[o * 4 + k] * seW[c * 68 + 64 + k];
    }
    s2l[p] = v;
    s2ws[(size_t)n * 1024 + p] = v;
  }
  __syncthreads();
  for (int p = t; p < 1024; p += 256) {
    int o = p >> 6, c = p & 63;
    float v = dB1[c];
    for (int k = 0; k < 64; ++k) v += s2l[o * 64 + k] * dW1[c * 64 + k];
    h1[p] = fmaxf(v, 0.f);
  }
  __syncthreads();
  if (t < 48) {
    int o = t / 3, dd = t % 3;
    float v = dB2[dd];
    for (int k = 0; k < 64; ++k) v += h1[o * 64 + k] * dW2[dd * 64 + k];
    lg[t] = v;
  }
  __syncthreads();
  if (t < 16) {
    float a = lg[t * 3], b = lg[t * 3 + 1], c = lg[t * 3 + 2];
    float m = fmaxf(a, fmaxf(b, c));
    float ea = expf(a - m), eb = expf(b - m), ec = expf(c - m);
    float s3 = ea + eb + ec;
    dyntws[(size_t)n * 48 + t * 3 + 0] = ea / s3;
    dyntws[(size_t)n * 48 + t * 3 + 1] = eb / s3;
    dyntws[(size_t)n * 48 + t * 3 + 2] = ec / s3;
  }
}

// ---------------------------------------------------------------------------
// K2: main interaction-network kernel, one workgroup per batch element
// ---------------------------------------------------------------------------
__global__ __launch_bounds__(512) void k2_main(
    const _Float16* __restrict__ wh,
    const float* __restrict__ s2ws, const float* __restrict__ dyntws,
    const float* __restrict__ relW1f, const float* __restrict__ relB1,
    const float* __restrict__ relB2,  const float* __restrict__ relB3,
    const float* __restrict__ attW1f, const float* __restrict__ attB1,
    const float* __restrict__ attB2,  const float* __restrict__ attW3f,
    const float* __restrict__ attB3,
    const float* __restrict__ selfB1, const float* __restrict__ selfB2,
    const float* __restrict__ affB1,  const float* __restrict__ affB2,
    const float* __restrict__ affB3,
    const float* __restrict__ outB1,  const float* __restrict__ outB2,
    float* __restrict__ dplws, float* __restrict__ result) {
  extern __shared__ char smem[];
  _Float16* comb = (_Float16*)(smem + L_COMB);
  _Float16* buf1 = (_Float16*)(smem + L_BUF1);
  _Float16* buf2 = (_Float16*)(smem + L_BUF2);
  _Float16* buf3 = (_Float16*)(smem + L_BUF3);
  _Float16* wst  = (_Float16*)(smem + L_WST);
  _Float16* s2h  = (_Float16*)(smem + L_S2H);
  float* distA   = (float*)(smem + L_DIST);
  float* attv    = (float*)(smem + L_ATT);
  float* reldyn  = (float*)(smem + L_RELD);
  _Float16* h1h  = (_Float16*)(smem + L_H1H);
  float* hf      = (float*)(smem + L_HF);
  _Float16* dph  = (_Float16*)(smem + L_DPH);
  float* dpf     = (float*)(smem + L_DPF);
  _Float16* f1h  = (_Float16*)(smem + L_F1H);
  float* f1f     = (float*)(smem + L_F1F);
  _Float16* f2h  = (_Float16*)(smem + L_F2H);
  _Float16* outa = (_Float16*)(smem + L_OUTA);
  _Float16* o1h  = (_Float16*)(smem + L_O1H);
  float* o1f     = (float*)(smem + L_O1F);
  float* res     = (float*)(smem + L_RES);
  float* dyntl   = (float*)(smem + L_DYNT);
  float* px      = (float*)(smem + L_PX);
  float* py      = (float*)(smem + L_PY);

  const int t = threadIdx.x;
  const int lane = t & 31;
  const int wv = t >> 5;           // 0..15
  const int lr = lane & 15;
  const int lh = (lane >> 4) & 1;
  const int mh = lh * 8;
  const int n = blockIdx.x;

  // ---- load s2, positions, dyn_types; build comb/dist; init res & outa tail
  for (int p = t; p < 1024; p += 512) s2h[p] = (_Float16)s2ws[(size_t)n * 1024 + p];
  if (t < 16) {
    px[t] = s2ws[(size_t)n * 1024 + t * 64 + 0];
    py[t] = s2ws[(size_t)n * 1024 + t * 64 + 1];
  }
  if (t < 48) dyntl[t] = dyntws[(size_t)n * 48 + t];
  __syncthreads();
  for (int p = t; p < 16384; p += 512) {
    int e = p >> 6, c = p & 63;
    comb[e * 128 + c]      = s2h[(e >> 4) * 64 + c];  // a1 = s2[i]
    comb[e * 128 + 64 + c] = s2h[(e & 15) * 64 + c];  // a2 = s2[j]
  }
  for (int e = t; e < 256; e += 512) {
    int i = e >> 4, j = e & 15;
    float dx = px[i] - px[j], dy = py[i] - py[j];
    distA[e] = dx * dx + dy * dy;
  }
  for (int p = t; p < 2048; p += 512) {
    int i = p >> 7, c = p & 127;
    res[p] = 0.f;
    if (c >= 64) outa[p] = s2h[i * 64 + c - 64];  // concat tail = s2, constant over d
  }
  __syncthreads();

  for (int dd = 0; dd < 3; ++dd) {
    const _Float16* whd = wh + dd * PERD;
    __builtin_prefetch((const void*)whd, 0, 1);  // global_prefetch_b8

    // ===== rel chain: comb -> r1 -> r2 -> rel3 =====
    stageW(wst, whd + OFF_RELW1, 16384, t); __syncthreads();
    edge_gemm1(comb, wst, relB1 + dd * 128, relW1f + (size_t)dd * 128 * 129,
               distA, buf1, wv, lr, lh, mh);
    __syncthreads();
    stageW(wst, whd + OFF_RELW2, 8192, t); __syncthreads();
    edge_gemm2(buf1, wst, relB2 + dd * 64, buf2, wv, lr, lh, mh);
    __syncthreads();
    stageW(wst, whd + OFF_RELW3, 4096, t); __syncthreads();
    edge_gemm3_res(buf2, wst, relB3 + dd * 64, buf3, wv, lr, lh, mh);
    __syncthreads();

    // ===== att chain: comb -> t1 -> t2 -> sigmoid scalar =====
    stageW(wst, whd + OFF_ATTW1, 16384, t); __syncthreads();
    edge_gemm1(comb, wst, attB1 + dd * 128, attW1f + (size_t)dd * 128 * 129,
               distA, buf1, wv, lr, lh, mh);
    __syncthreads();
    stageW(wst, whd + OFF_ATTW2, 8192, t); __syncthreads();
    edge_gemm2(buf1, wst, attB2 + dd * 64, buf2, wv, lr, lh, mh);
    __syncthreads();
    {
      const float* w3 = attW3f + dd * 64;
      float b3 = attB3[dd];
      for (int e = t; e < 256; e += 512) {
        float v = b3;
        for (int k = 0; k < 64; ++k) v += (float)buf2[e * 64 + k] * w3[k];
        int i = e >> 4, j = e & 15;
        attv[e] = (i == j) ? 0.f : 1.f / (1.f + expf(-v));  // fold diag mask in
      }
    }
    __syncthreads();
    // rel_dyn[i,c] = sum_j att * rel3
    for (int p = t; p < 1024; p += 512) {
      int i = p >> 6, c = p & 63;
      float acc = 0.f;
      for (int j = 0; j < 16; ++j)
        acc += attv[i * 16 + j] * (float)buf3[(i * 16 + j) * 64 + c];
      reldyn[p] = acc;
    }
    __syncthreads();

    // ===== node chain (single 16-row M-tile, WMMA w/ global B-frags) =====
    if (wv < 4) {  // self h1 = relu(lin(s2))
      v8f c = node_gemm(s2h, 64, whd + OFF_SELFW1, 64, selfB1 + dd * 64, wv, 2, lr, lh);
#pragma unroll
      for (int r = 0; r < 8; ++r) {
        int row = mh + r, col = wv * 16 + lr;
        float v = fmaxf(c[r], 0.f);
        h1h[row * 64 + col] = (_Float16)v;
        hf[row * 64 + col] = v;
      }
    }
    __syncthreads();
    if (wv < 4) {  // dyn_pred = lin(h1) + h1 + rel_dyn
      v8f c = node_gemm(h1h, 64, whd + OFF_SELFW2, 64, selfB2 + dd * 64, wv, 2, lr, lh);
#pragma unroll
      for (int r = 0; r < 8; ++r) {
        int row = mh + r, col = wv * 16 + lr;
        float v = c[r] + hf[row * 64 + col] + reldyn[row * 64 + col];
        dpf[row * 64 + col] = v;
        dph[row * 64 + col] = (_Float16)v;
      }
    }
    __syncthreads();
    if (dd == 2)  // reward head consumes last dyn_pred
      for (int p = t; p < 1024; p += 512) dplws[(size_t)n * 1024 + p] = dpf[p];
    if (wv < 4) {  // f1 = tanh(lin(dyn_pred))
      v8f c = node_gemm(dph, 64, whd + OFF_AFFW1, 64, affB1 + dd * 64, wv, 2, lr, lh);
#pragma unroll
      for (int r = 0; r < 8; ++r) {
        int row = mh + r, col = wv * 16 + lr;
        float v = tanhf(c[r]);
        f1h[row * 64 + col] = (_Float16)v;
        f1f[row * 64 + col] = v;
      }
    }
    __syncthreads();
    if (wv < 4) {  // f2 = tanh(lin(f1)) + f1
      v8f c = node_gemm(f1h, 64, whd + OFF_AFFW2, 64, affB2 + dd * 64, wv, 2, lr, lh);
#pragma unroll
      for (int r = 0; r < 8; ++r) {
        int row = mh + r, col = wv * 16 + lr;
        float v = tanhf(c[r]) + f1f[row * 64 + col];
        f2h[row * 64 + col] = (_Float16)v;
      }
    }
    __syncthreads();
    if (wv < 4) {  // f3 = lin(f2) -> first half of concat buffer
      v8f c = node_gemm(f2h, 64, whd + OFF_AFFW3, 64, affB3 + dd * 64, wv, 2, lr, lh);
#pragma unroll
      for (int r = 0; r < 8; ++r) {
        int row = mh + r, col = wv * 16 + lr;
        outa[row * 128 + col] = (_Float16)c[r];
      }
    }
    __syncthreads();
    if (wv < 8) {  // o1 = tanh(lin([f3, s2]))
      v8f c = node_gemm(outa, 128, whd + OFF_OUTW1, 128, outB1 + dd * 128, wv, 4, lr, lh);
#pragma unroll
      for (int r = 0; r < 8; ++r) {
        int row = mh + r, col = wv * 16 + lr;
        float v = tanhf(c[r]);
        o1h[row * 128 + col] = (_Float16)v;
        o1f[row * 128 + col] = v;
      }
    }
    __syncthreads();
    if (wv < 8) {  // result += (lin(o1) + o1) * dyn_types[:, d]
      v8f c = node_gemm(o1h, 128, whd + OFF_OUTW2, 128, outB2 + dd * 128, wv, 4, lr, lh);
#pragma unroll
      for (int r = 0; r < 8; ++r) {
        int row = mh + r, col = wv * 16 + lr;
        res[row * 128 + col] += (c[r] + o1f[row * 128 + col]) * dyntl[row * 3 + dd];
      }
    }
    __syncthreads();
  }

  for (int p = t; p < 2048; p += 512) result[(size_t)n * 2048 + p] = res[p];
}

// ---------------------------------------------------------------------------
// K3: reward head (tiny, f32 VALU)
// ---------------------------------------------------------------------------
__global__ __launch_bounds__(64) void k3_reward(
    const float* __restrict__ dpl,
    const float* __restrict__ W1, const float* __restrict__ b1,
    const float* __restrict__ W2, const float* __restrict__ b2,
    const float* __restrict__ rW1, const float* __restrict__ rb1,
    const float* __restrict__ rW2, const float* __restrict__ rb2,
    const float* __restrict__ rW3, const float* __restrict__ rb3,
    float* __restrict__ outRw) {
  __shared__ float dp[1024], h[64], acc[64], t1[32], t2[16];
  int n = blockIdx.x, t = threadIdx.x;
  for (int p = t; p < 1024; p += 64) dp[p] = dpl[(size_t)n * 1024 + p];
  acc[t] = 0.f;
  __syncthreads();
  for (int o = 0; o < 16; ++o) {
    float v = b1[t];
    for (int k = 0; k < 64; ++k) v += dp[o * 64 + k] * W1[t * 64 + k];
    h[t] = fmaxf(v, 0.f);
    __syncthreads();
    float u = b2[t];
    for (int k = 0; k < 64; ++k) u += h[k] * W2[t * 64 + k];
    acc[t] += u;
    __syncthreads();
  }
  if (t < 32) {
    float v = rb1[t];
    for (int k = 0; k < 64; ++k) v += acc[k] * rW1[t * 64 + k];
    t1[t] = fmaxf(v, 0.f);
  }
  __syncthreads();
  if (t < 16) {
    float v = rb2[t];
    for (int k = 0; k < 32; ++k) v += t1[k] * rW2[t * 32 + k];
    t2[t] = fmaxf(v, 0.f);
  }
  __syncthreads();
  if (t == 0) {
    float v = rb3[0];
    for (int k = 0; k < 16; ++k) v += t2[k] * rW3[k];
    outRw[n] = 1.f / (1.f + expf(-v));
  }
}

// ---------------------------------------------------------------------------
// Host launcher
// ---------------------------------------------------------------------------
extern "C" void kernel_launch(void* const* d_in, const int* in_sizes, int n_in,
                              void* d_out, int out_size, void* d_ws, size_t ws_size,
                              hipStream_t stream) {
  (void)in_sizes; (void)n_in; (void)out_size; (void)ws_size;
  const float* s       = (const float*)d_in[0];
  const float* actions = (const float*)d_in[1];
  const float* actW  = (const float*)d_in[2];
  const float* actB  = (const float*)d_in[3];
  const float* seW   = (const float*)d_in[4];
  const float* seB   = (const float*)d_in[5];
  const float* dW1   = (const float*)d_in[6];
  const float* dB1   = (const float*)d_in[7];
  const float* dW2   = (const float*)d_in[8];
  const float* dB2   = (const float*)d_in[9];
  const float* selfW1 = (const float*)d_in[10];
  const float* selfB1 = (const float*)d_in[11];
  const float* selfW2 = (const float*)d_in[12];
  const float* selfB2 = (const float*)d_in[13];
  const float* relW1 = (const float*)d_in[14];
  const float* relB1 = (const float*)d_in[15];
  const float* relW2 = (const float*)d_in[16];
  const float* relB2 = (const float*)d_in[17];
  const float* relW3 = (const float*)d_in[18];
  const float* relB3 = (const float*)d_in[19];
  const float* attW1 = (const float*)d_in[20];
  const float* attB1 = (const float*)d_in[21];
  const float* attW2 = (const float*)d_in[22];
  const float* attB2 = (const float*)d_in[23];
  const float* attW3 = (const float*)d_in[24];
  const float* attB3 = (const float*)d_in[25];
  const float* affW1 = (const float*)d_in[26];
  const float* affB1 = (const float*)d_in[27];
  const float* affW2 = (const float*)d_in[28];
  const float* affB2 = (const float*)d_in[29];
  const float* affW3 = (const float*)d_in[30];
  const float* affB3 = (const float*)d_in[31];
  const float* outW1 = (const float*)d_in[32];
  const float* outB1 = (const float*)d_in[33];
  const float* outW2 = (const float*)d_in[34];
  const float* outB2 = (const float*)d_in[35];
  const float* rw0W1 = (const float*)d_in[36];
  const float* rw0B1 = (const float*)d_in[37];
  const float* rw0W2 = (const float*)d_in[38];
  const float* rw0B2 = (const float*)d_in[39];
  const float* rw1W1 = (const float*)d_in[40];
  const float* rw1B1 = (const float*)d_in[41];
  const float* rw1W2 = (const float*)d_in[42];
  const float* rw1B2 = (const float*)d_in[43];
  const float* rw1W3 = (const float*)d_in[44];
  const float* rw1B3 = (const float*)d_in[45];

  char* ws = (char*)d_ws;
  float* s2ws   = (float*)(ws + WS_S2);
  float* dyntws = (float*)(ws + WS_DYNT);
  float* dplws  = (float*)(ws + WS_DPL);
  _Float16* wh  = (_Float16*)(ws + WS_WH);
  float* result = (float*)d_out;
  float* reward = result + (size_t)NB * OO * 128;

  k0_convert<<<(3 * PERD + 255) / 256, 256, 0, stream>>>(
      wh, relW1, attW1, relW2, attW2, relW3, selfW1, selfW2,
      affW1, affW2, affW3, outW1, outW2);
  k1_node<<<NB, 256, 0, stream>>>(s, actions, actW, actB, seW, seB,
                                  dW1, dB1, dW2, dB2, s2ws, dyntws);
  k2_main<<<NB, 512, SMEM_TOTAL, stream>>>(
      wh, s2ws, dyntws, relW1, relB1, relB2, relB3, attW1, attB1, attB2,
      attW3, attB3, selfB1, selfB2, affB1, affB2, affB3, outB1, outB2,
      dplws, result);
  k3_reward<<<NB, 64, 0, stream>>>(dplws, rw0W1, rw0B1, rw0W2, rw0B2,
                                   rw1W1, rw1B1, rw1W2, rw1B2, rw1W3, rw1B3,
                                   reward);
}
